// GAT_26525718020641
// MI455X (gfx1250) — matrix operand order repaired and checked by reading, not compiled, not run on hardware
//
#include <hip/hip_runtime.h>
#include <hip/hip_fp16.h>

typedef __attribute__((ext_vector_type(16))) _Float16 v16h;
typedef __attribute__((ext_vector_type(8)))  _Float16 v8h;
typedef __attribute__((ext_vector_type(8)))  float    v8f;

#define N_NODES   50000
#define N_EDGES   800000
#define IN_DIM    128
#define HID       64
#define HEADS     4
#define OUT_DIM   20
#define NUM_GRAPHS 64
#define E_TOT     (N_EDGES + N_NODES)
#define NEG_SLOPE 0.2f

// ---------------- small helpers ----------------

__device__ __forceinline__ float atomic_max_f32(float* addr, float v) {
  // sign-split trick: works with -inf initialization
  if (v >= 0.0f)
    return __int_as_float(atomicMax((int*)addr, __float_as_int(v)));
  else
    return __uint_as_float(atomicMin((unsigned int*)addr, __float_as_uint(v)));
}

__global__ void fill_f32(float* __restrict__ p, float v, int n) {
  int i = blockIdx.x * blockDim.x + threadIdx.x;
  if (i < n) p[i] = v;
}

__global__ void f32_to_f16(const float* __restrict__ in, _Float16* __restrict__ out, int n) {
  int i = blockIdx.x * blockDim.x + threadIdx.x;
  if (i < n) out[i] = (_Float16)in[i];
}

// ---------------- WMMA GEMM: C[M,N] = A[M,K] * W[N,K]^T ----------------
// One wave32 computes a 16x16 tile, accumulating over K in steps of 32 via
// v_wmma_f32_16x16x32_f16. M%16==0, N%16==0, K%32==0 (true for this net).
__global__ void wmma_gemm_wT(const _Float16* __restrict__ A,
                             const _Float16* __restrict__ W,
                             float* __restrict__ C,
                             int M, int N, int K) {
  int wave = (blockIdx.x * blockDim.x + threadIdx.x) >> 5;
  int lane = threadIdx.x & 31;
  int Nt = N >> 4;
  int Mt = M >> 4;
  if (wave >= Mt * Nt) return;
  int mt = wave / Nt;
  int nt = wave % Nt;

  int arow = mt * 16 + (lane & 15);          // A-matrix: lane -> row M
  int brow = nt * 16 + (lane & 15);          // B = W^T: lane -> column N (= W row)
  int kbA  = (lane < 16) ? 0 : 16;           // contiguous K order, matches B
  int kbB  = (lane < 16) ? 0 : 16;           // contiguous K order, matches A

  const _Float16* aptr = A + (size_t)arow * K;
  const _Float16* bptr = W + (size_t)brow * K;

  v8f acc = {};
  for (int kk = 0; kk < K; kk += 32) {
    // A lane data: halves j=0..15 -> K = kk+kbA+j
    v8h alo = *(const v8h*)(aptr + kk + kbA);
    v8h ahi = *(const v8h*)(aptr + kk + kbA + 8);
    // B lane data: halves j=0..15 -> K = kk+kbB+j at column brow
    v16h b = *(const v16h*)(bptr + kk + kbB);
    v16h a;
#pragma unroll
    for (int i = 0; i < 8; ++i) { a[i] = alo[i]; a[i + 8] = ahi[i]; }
    acc = __builtin_amdgcn_wmma_f32_16x16x32_f16(
        /*neg_a=*/false, a, /*neg_b=*/false, b,
        /*c_mod=*/(short)0, acc, /*reuse_a=*/false, /*reuse_b=*/false);
  }

  // C/D layout: VGPR r -> M = r (lanes 0-15) or 8+r (lanes 16-31); N = lane&15
  int ncol  = nt * 16 + (lane & 15);
  int rbase = mt * 16 + ((lane < 16) ? 0 : 8);
#pragma unroll
  for (int r = 0; r < 8; ++r)
    C[(size_t)(rbase + r) * N + ncol] = acc[r];
}

// ---------------- attention coefficients: alpha_s/alpha_d per (node, head) ----
__global__ void attn_coeff(const float* __restrict__ h,      // [N, H*C]
                           const float* __restrict__ a_src,  // [H*C]
                           const float* __restrict__ a_dst,  // [H*C]
                           float* __restrict__ as_out,       // [N, H]
                           float* __restrict__ ad_out,       // [N, H]
                           int n_nodes, int H, int Cdim) {
  int i = blockIdx.x * blockDim.x + threadIdx.x;
  if (i >= n_nodes * H) return;
  int n = i / H, hd = i % H;
  const float* hp  = h + (size_t)n * H * Cdim + hd * Cdim;
  const float* asv = a_src + hd * Cdim;
  const float* adv = a_dst + hd * Cdim;
  float s = 0.f, d = 0.f;
  for (int c = 0; c < Cdim; ++c) { float v = hp[c]; s += v * asv[c]; d += v * adv[c]; }
  as_out[i] = s;
  ad_out[i] = d;
}

// ---------------- edge pass 1: segment max of leaky_relu(as[src]+ad[dst]) ----
__global__ void edge_max(const int* __restrict__ ei,   // [2, N_EDGES]
                         const float* __restrict__ as, const float* __restrict__ ad,
                         float* __restrict__ emax, int H) {
  int i = blockIdx.x * blockDim.x + threadIdx.x;
  if (i >= E_TOT * H) return;
  int e = i / H, hd = i % H;
  int s, d;
  if (e < N_EDGES) { s = ei[e]; d = ei[N_EDGES + e]; }
  else             { s = d = e - N_EDGES; }            // self-loops
  float v = as[s * H + hd] + ad[d * H + hd];
  v = (v > 0.f) ? v : NEG_SLOPE * v;
  atomic_max_f32(&emax[d * H + hd], v);
}

// ---------------- edge pass 2: scatter w*h[src] and denom (wave per edge) ----
__global__ void edge_scatter(const int* __restrict__ ei,
                             const float* __restrict__ as, const float* __restrict__ ad,
                             const float* __restrict__ emax,
                             const float* __restrict__ h,       // [N, H*C]
                             float* __restrict__ denom,         // [N, H]
                             float* __restrict__ accum,         // [N, H*C]
                             int H, int Cdim) {
  int wave = (blockIdx.x * blockDim.x + threadIdx.x) >> 5;
  int lane = threadIdx.x & 31;
  if (wave >= E_TOT) return;
  int e = wave;
  int s, d;
  if (e < N_EDGES) { s = ei[e]; d = ei[N_EDGES + e]; }
  else             { s = d = e - N_EDGES; }
  int HC = H * Cdim;
  const float* hs   = h + (size_t)s * HC;
  float*       accd = accum + (size_t)d * HC;
  for (int hd = 0; hd < H; ++hd) {
    float v = as[s * H + hd] + ad[d * H + hd];
    v = (v > 0.f) ? v : NEG_SLOPE * v;
    float w = __expf(v - emax[d * H + hd]);
    if (lane == 0) atomicAdd(&denom[d * H + hd], w);
    for (int c = lane; c < Cdim; c += 32)
      atomicAdd(&accd[hd * Cdim + c], w * hs[hd * Cdim + c]);
  }
}

// ---------------- finalize: normalize, +bias, ELU; write f32 and/or f16 ------
__global__ void finalize_elu(const float* __restrict__ accum,
                             const float* __restrict__ denom,
                             const float* __restrict__ bias,
                             float* __restrict__ outf,      // may be null
                             _Float16* __restrict__ outh,   // may be null
                             int n_nodes, int H, int Cdim) {
  int HC = H * Cdim;
  int i = blockIdx.x * blockDim.x + threadIdx.x;
  if (i >= n_nodes * HC) return;
  int n = i / HC, j = i % HC, hd = j / Cdim;
  float v = accum[i] / (denom[n * H + hd] + 1e-16f) + bias[j];
  v = (v > 0.f) ? v : (__expf(v) - 1.0f);                 // ELU
  if (outf) outf[i] = v;
  if (outh) outh[i] = (_Float16)v;
}

// ---------------- graph pooling: segment max over batch ----------------------
__global__ void pool_max(const float* __restrict__ h,     // [N, HID]
                         const int* __restrict__ batch,
                         float* __restrict__ pooled,      // [G, HID], init -inf
                         int n_nodes, int Cdim) {
  int i = blockIdx.x * blockDim.x + threadIdx.x;
  if (i >= n_nodes * Cdim) return;
  int n = i / Cdim, c = i % Cdim;
  atomic_max_f32(&pooled[batch[n] * Cdim + c], h[i]);
}

// ---------------- tiny FC head ----------------------------------------------
__global__ void fc_head(const float* __restrict__ pooled,  // [G, HID]
                        const float* __restrict__ Wf,      // [OUT, HID]
                        const float* __restrict__ bf,      // [OUT]
                        float* __restrict__ out) {         // [G, OUT]
  int i = blockIdx.x * blockDim.x + threadIdx.x;
  if (i >= NUM_GRAPHS * OUT_DIM) return;
  int g = i / OUT_DIM, o = i % OUT_DIM;
  float s = bf[o];
  for (int c = 0; c < HID; ++c) s += pooled[g * HID + c] * Wf[o * HID + c];
  out[i] = s;
}

// =============================================================================

static inline int cdiv(long long a, long long b) { return (int)((a + b - 1) / b); }

extern "C" void kernel_launch(void* const* d_in, const int* in_sizes, int n_in,
                              void* d_out, int out_size, void* d_ws, size_t ws_size,
                              hipStream_t stream) {
  (void)in_sizes; (void)n_in; (void)out_size; (void)ws_size;
  const float* x      = (const float*)d_in[0];
  const int*   ei     = (const int*)  d_in[1];
  const int*   batch  = (const int*)  d_in[2];
  const float* W1     = (const float*)d_in[3];
  const float* a_src1 = (const float*)d_in[4];
  const float* a_dst1 = (const float*)d_in[5];
  const float* b1     = (const float*)d_in[6];
  const float* W2     = (const float*)d_in[7];
  const float* a_src2 = (const float*)d_in[8];
  const float* a_dst2 = (const float*)d_in[9];
  const float* b2     = (const float*)d_in[10];
  const float* fcW    = (const float*)d_in[11];
  const float* fcb    = (const float*)d_in[12];
  float* out = (float*)d_out;

  // bump allocator over workspace (all offsets 256B aligned)
  char* p = (char*)d_ws;
  auto alloc = [&](size_t bytes) -> char* {
    char* r = p;
    p += (bytes + 255) & ~(size_t)255;
    return r;
  };
  const int HC1 = HEADS * HID;                      // 256
  _Float16* xh     = (_Float16*)alloc((size_t)N_NODES * IN_DIM * 2);
  _Float16* W1h    = (_Float16*)alloc((size_t)HC1 * IN_DIM * 2);
  _Float16* W2h    = (_Float16*)alloc((size_t)HID * HC1 * 2);
  float*    h1     = (float*)alloc((size_t)N_NODES * HC1 * 4);
  float*    as1    = (float*)alloc((size_t)N_NODES * HEADS * 4);
  float*    ad1    = (float*)alloc((size_t)N_NODES * HEADS * 4);
  float*    emax1  = (float*)alloc((size_t)N_NODES * HEADS * 4);
  float*    den1   = (float*)alloc((size_t)N_NODES * HEADS * 4);
  float*    acc1   = (float*)alloc((size_t)N_NODES * HC1 * 4);
  _Float16* hact1h = (_Float16*)alloc((size_t)N_NODES * HC1 * 2);
  float*    h2     = (float*)alloc((size_t)N_NODES * HID * 4);
  float*    as2    = (float*)alloc((size_t)N_NODES * 4);
  float*    ad2    = (float*)alloc((size_t)N_NODES * 4);
  float*    emax2  = (float*)alloc((size_t)N_NODES * 4);
  float*    den2   = (float*)alloc((size_t)N_NODES * 4);
  float*    acc2   = (float*)alloc((size_t)N_NODES * HID * 4);
  float*    hact2  = (float*)alloc((size_t)N_NODES * HID * 4);
  float*    pooled = (float*)alloc((size_t)NUM_GRAPHS * HID * 4);

  const int TB = 256;
  const float NEG_INF = -__builtin_inff();

  // --- conv1 linear (WMMA) ---
  f32_to_f16<<<cdiv((long long)N_NODES * IN_DIM, TB), TB, 0, stream>>>(x, xh, N_NODES * IN_DIM);
  f32_to_f16<<<cdiv(HC1 * IN_DIM, TB), TB, 0, stream>>>(W1, W1h, HC1 * IN_DIM);
  f32_to_f16<<<cdiv(HID * HC1, TB), TB, 0, stream>>>(W2, W2h, HID * HC1);
  {
    int tiles = (N_NODES / 16) * (HC1 / 16);
    wmma_gemm_wT<<<cdiv((long long)tiles * 32, TB), TB, 0, stream>>>(xh, W1h, h1, N_NODES, HC1, IN_DIM);
  }
  attn_coeff<<<cdiv(N_NODES * HEADS, TB), TB, 0, stream>>>(h1, a_src1, a_dst1, as1, ad1, N_NODES, HEADS, HID);

  // --- conv1 edge phase ---
  fill_f32<<<cdiv(N_NODES * HEADS, TB), TB, 0, stream>>>(emax1, NEG_INF, N_NODES * HEADS);
  fill_f32<<<cdiv(N_NODES * HEADS, TB), TB, 0, stream>>>(den1, 0.f, N_NODES * HEADS);
  fill_f32<<<cdiv((long long)N_NODES * HC1, TB), TB, 0, stream>>>(acc1, 0.f, N_NODES * HC1);
  edge_max<<<cdiv((long long)E_TOT * HEADS, TB), TB, 0, stream>>>(ei, as1, ad1, emax1, HEADS);
  edge_scatter<<<cdiv((long long)E_TOT * 32, TB), TB, 0, stream>>>(ei, as1, ad1, emax1, h1, den1, acc1, HEADS, HID);
  finalize_elu<<<cdiv((long long)N_NODES * HC1, TB), TB, 0, stream>>>(acc1, den1, b1, (float*)nullptr, hact1h,
                                                                     N_NODES, HEADS, HID);

  // --- conv2 linear (WMMA) ---
  {
    int tiles = (N_NODES / 16) * (HID / 16);
    wmma_gemm_wT<<<cdiv((long long)tiles * 32, TB), TB, 0, stream>>>(hact1h, W2h, h2, N_NODES, HID, HC1);
  }
  attn_coeff<<<cdiv(N_NODES, TB), TB, 0, stream>>>(h2, a_src2, a_dst2, as2, ad2, N_NODES, 1, HID);

  // --- conv2 edge phase ---
  fill_f32<<<cdiv(N_NODES, TB), TB, 0, stream>>>(emax2, NEG_INF, N_NODES);
  fill_f32<<<cdiv(N_NODES, TB), TB, 0, stream>>>(den2, 0.f, N_NODES);
  fill_f32<<<cdiv((long long)N_NODES * HID, TB), TB, 0, stream>>>(acc2, 0.f, N_NODES * HID);
  edge_max<<<cdiv(E_TOT, TB), TB, 0, stream>>>(ei, as2, ad2, emax2, 1);
  edge_scatter<<<cdiv((long long)E_TOT * 32, TB), TB, 0, stream>>>(ei, as2, ad2, emax2, h2, den2, acc2, 1, HID);
  finalize_elu<<<cdiv((long long)N_NODES * HID, TB), TB, 0, stream>>>(acc2, den2, b2, hact2, (_Float16*)nullptr,
                                                                     N_NODES, 1, HID);

  // --- pooling + FC head ---
  fill_f32<<<cdiv(NUM_GRAPHS * HID, TB), TB, 0, stream>>>(pooled, NEG_INF, NUM_GRAPHS * HID);
  pool_max<<<cdiv((long long)N_NODES * HID, TB), TB, 0, stream>>>(hact2, batch, pooled, N_NODES, HID);
  fc_head<<<cdiv(NUM_GRAPHS * OUT_DIM, TB), TB, 0, stream>>>(pooled, fcW, fcb, out);
}
